// GAT_53455162966158
// MI455X (gfx1250) — matrix operand order repaired and checked by reading, not compiled, not run on hardware
//
#include <hip/hip_runtime.h>
#include <hip/hip_bf16.h>

typedef _Float16 half_t;
typedef __attribute__((ext_vector_type(16))) _Float16 v16h;
typedef __attribute__((ext_vector_type(8)))  _Float16 v8h;
typedef __attribute__((ext_vector_type(8)))  float    v8f;
typedef __attribute__((ext_vector_type(4)))  int      v4i;

#define B_DIM 4
#define L_DIM 2048
#define DIN   256
#define DH    512
#define H_DIM 8
#define E_DIM 64
#define DCAT  1024
#define M_ROWS (B_DIM * L_DIM)   // 8192
#define LW    (L_DIM / 32)       // mask words per row

__device__ __constant__ float kScale = 0.08838834764831845f; // 1/sqrt(128)
#define NEG_BIG (-9.0e15f)

#if __has_builtin(__builtin_amdgcn_sched_barrier)
#define SCHED_FENCE() __builtin_amdgcn_sched_barrier(0)
#else
#define SCHED_FENCE()
#endif

// ---------------------------------------------------------------------------
// WMMA fragment helpers (wave32, v_wmma_f32_16x16x32_f16 layouts per CDNA5 ISA)
// ---------------------------------------------------------------------------
__device__ inline v16h make_v16h(v8h lo, v8h hi) {
  v16h r;
#pragma unroll
  for (int i = 0; i < 8; i++) { r[i] = lo[i]; r[i + 8] = hi[i]; }
  return r;
}

// A-matrix 16x32 f16: lane = row (m = lane%16), group g = lane/16.
// halfs 0..7  -> K = g*8 + 0..7 ; halfs 8..15 -> K = 16 + g*8 + 0..7
__device__ inline v16h load_a_frag(const half_t* __restrict__ src, int ld,
                                   int row0, int k0) {
  int lane = threadIdx.x & 31;
  int m = lane & 15;
  int g = lane >> 4;
  const half_t* p = src + (long)(row0 + m) * ld + k0;
  v8h lo = *(const v8h*)(p + g * 8);
  v8h hi = *(const v8h*)(p + 16 + g * 8);
  return make_v16h(lo, hi);
}

// B-matrix 32x16 f16, operand supplied TRANSPOSED: srcT is [N][K] row-major.
// lane = column (n = lane%16), group g = lane/16; halfs 0..7 -> K = g*8 + 0..7 ; halfs 8..15 -> K = 16 + g*8 + 0..7
__device__ inline v16h load_b_frag(const half_t* __restrict__ srcT, int ld,
                                   int n0, int k0) {
  int lane = threadIdx.x & 31;
  int n = lane & 15;
  int g = lane >> 4;
  const half_t* p = srcT + (long)(n0 + n) * ld + k0;
  v8h lo = *(const v8h*)(p + g * 8);
  v8h hi = *(const v8h*)(p + 16 + g * 8);
  return make_v16h(lo, hi);
}

__device__ inline v8f wmma16(v16h a, v16h b, v8f c) {
  return __builtin_amdgcn_wmma_f32_16x16x32_f16(
      /*neg_a=*/false, a, /*neg_b=*/false, b,
      /*c_mod=*/(short)0, c, /*reuse_a=*/false, /*reuse_b=*/false);
}

// ---------------------------------------------------------------------------
// Conversion / prep kernels
// ---------------------------------------------------------------------------
__global__ void cvt_f32_to_f16(const float* __restrict__ in,
                               half_t* __restrict__ out, int n) {
  int i = blockIdx.x * blockDim.x + threadIdx.x;
  if (i < n) out[i] = (half_t)in[i];
}

// W is [K][N] f32 row-major; WT is [N][K] f16 row-major
__global__ void cvt_transpose(const float* __restrict__ W,
                              half_t* __restrict__ WT, int K, int N) {
  int i = blockIdx.x * blockDim.x + threadIdx.x;
  if (i < K * N) {
    int k = i / N, n = i - k * N;
    WT[(long)n * K + k] = (half_t)W[i];
  }
}

// bit-pack Gmat: word i covers Gmat[i*32 .. i*32+31] (row-major)
__global__ void pack_mask(const int* __restrict__ G,
                          unsigned* __restrict__ Gb, int nwords) {
  int i = blockIdx.x * blockDim.x + threadIdx.x;
  if (i < nwords) {
    const v4i* p = (const v4i*)(G + (long)i * 32);
    unsigned m = 0;
#pragma unroll
    for (int q = 0; q < 8; q++) {
      v4i v = p[q];
#pragma unroll
      for (int j = 0; j < 4; j++) m |= (v[j] > 0 ? 1u : 0u) << (q * 4 + j);
    }
    Gb[i] = m;
  }
}

// ---------------------------------------------------------------------------
// f16 WMMA GEMM: C = A[M,K] @ B[K,N] (+bias), B given transposed.
// Block = 256 threads (8 waves); wave tile = 16 rows x 128 cols.
// B tile staged through double-buffered LDS; next tile prefetched to regs.
// All 8 B fragments preloaded into distinct regs, sched-fenced before MACs.
// Grid: (M/128, N/128).
// ---------------------------------------------------------------------------
__global__ __launch_bounds__(256) void gemm_wmma_f16(
    const half_t* __restrict__ A, int lda,
    const half_t* __restrict__ BT, int ldb,
    const float* __restrict__ bias, int K,
    float* __restrict__ C32, int ldc32,
    half_t* __restrict__ C16, int ldc16,
    half_t* __restrict__ shT) {
  __shared__ __align__(16) half_t bt[2][128 * 32];

  int tid = threadIdx.x;
  int lane = tid & 31;
  int w = tid >> 5;
  int g = lane >> 4, n16 = lane & 15;
  int m0 = blockIdx.x * 128 + w * 16;
  int nb = blockIdx.y * 128;

  // staging: each thread owns 16 halfs of the 128x32 BT tile
  int srow = tid >> 1;        // 0..127 (column of C)
  int scol = (tid & 1) * 16;  // 0 or 16 (k within chunk)
  const half_t* bsrc = BT + (long)(nb + srow) * ldb + scol;

  v8f acc[8];
#pragma unroll
  for (int j = 0; j < 8; j++) acc[j] = (v8f)(0.0f);

  {  // prologue: stage k-chunk 0 into buffer 0
    v8h t0 = *(const v8h*)(bsrc);
    v8h t1 = *(const v8h*)(bsrc + 8);
    *(v8h*)&bt[0][srow * 32 + scol] = t0;
    *(v8h*)&bt[0][srow * 32 + scol + 8] = t1;
  }
  v16h a_cur = load_a_frag(A, lda, m0, 0);
  __syncthreads();

  for (int kk = 0; kk < K; kk += 32) {
    int cur = (kk >> 5) & 1;
    bool has_next = (kk + 32) < K;
    v8h t0, t1;
    v16h a_next;
    if (has_next) {  // issue next-tile global loads before the MACs
      t0 = *(const v8h*)(bsrc + kk + 32);
      t1 = *(const v8h*)(bsrc + kk + 40);
      a_next = load_a_frag(A, lda, m0, kk + 32);
    }
    // preload ALL B fragments (distinct regs), fence, then the 8 MACs
    v16h bfr[8];
#pragma unroll
    for (int j = 0; j < 8; j++)
      bfr[j] = load_b_frag(&bt[cur][0], 32, j * 16, 0);
    SCHED_FENCE();
#pragma unroll
    for (int j = 0; j < 8; j++) acc[j] = wmma16(a_cur, bfr[j], acc[j]);
    SCHED_FENCE();
    if (has_next) {
      *(v8h*)&bt[cur ^ 1][srow * 32 + scol] = t0;
      *(v8h*)&bt[cur ^ 1][srow * 32 + scol + 8] = t1;
      a_cur = a_next;
    }
    __syncthreads();
  }

#pragma unroll
  for (int j = 0; j < 8; j++) {
    int col = nb + j * 16 + n16;
    float bv = bias ? bias[col] : 0.0f;
#pragma unroll
    for (int r = 0; r < 8; r++) {
      int row = m0 + g * 8 + r;
      float v = acc[j][r] + bv;
      if (C32) C32[(long)row * ldc32 + col] = v;
      if (C16) C16[(long)row * ldc16 + col] = (half_t)v;
      if (shT) {  // head-transposed copy: shT[b][h][e][l]
        int bb = row >> 11, l = row & (L_DIM - 1);
        int hh = col >> 6, e = col & 63;
        shT[((long)((bb * H_DIM + hh) * E_DIM + e)) * L_DIM + l] = (half_t)v;
      }
    }
  }
}

// ---------------------------------------------------------------------------
// Fused flash attention. Block = 256 thr (8 waves), each wave owns 16 query
// rows. Grid: (L/128, H, B). Double-buffered K/vT tiles; packed bitmask.
// Fragment loads grouped + sched-fenced ahead of WMMAs.
// Writes s_out (f16) into concat[:,512:1024].
// ---------------------------------------------------------------------------
__global__ __launch_bounds__(256) void attn_wmma(
    const half_t* __restrict__ qf, const half_t* __restrict__ kf,
    const half_t* __restrict__ shT, const unsigned* __restrict__ Gbits,
    half_t* __restrict__ concatA) {
  __shared__ __align__(16) half_t lds_k[2][32 * 64];    // [key][e]
  __shared__ __align__(16) half_t lds_vT[2][64 * 32];   // [e][key]
  __shared__ __align__(16) half_t lds_p[8][16 * 32];    // per-wave P tile

  int tid = threadIdx.x;
  int lane = tid & 31, w = tid >> 5;
  int n16 = lane & 15, g = lane >> 4;
  int h = blockIdx.y, b = blockIdx.z;
  int q0 = blockIdx.x * 128 + w * 16;  // local query row base

  // q fragments (held in regs for the whole sweep)
  v16h qa0 = load_a_frag(qf, DH, b * L_DIM + q0, h * E_DIM);
  v16h qa1 = load_a_frag(qf, DH, b * L_DIM + q0, h * E_DIM + 32);

  float m_s[8], l_s[8];
  v8f o[4];
#pragma unroll
  for (int r = 0; r < 8; r++) { m_s[r] = -1.0e30f; l_s[r] = 0.0f; }
#pragma unroll
  for (int t = 0; t < 4; t++) o[t] = (v8f)(0.0f);

  const half_t* kbase = kf + (long)(b * L_DIM) * DH + h * E_DIM;
  const half_t* vTbase = shT + (long)((b * H_DIM + h) * E_DIM) * L_DIM;
  const float sc = kScale;

  // staging coords (one v8h each for K tile and vT tile)
  int kr = tid >> 3, kc = (tid & 7) * 8;  // K tile: [32][64]
  int ve = tid >> 2, vc = (tid & 3) * 8;  // vT tile: [64][32]
  const half_t* ksrc = kbase + (long)kr * DH + kc;
  const half_t* vsrc = vTbase + (long)ve * L_DIM + vc;

  {  // prologue: stage block 0
    v8h kv = *(const v8h*)(ksrc);
    v8h vv = *(const v8h*)(vsrc);
    *(v8h*)&lds_k[0][kr * 64 + kc] = kv;
    *(v8h*)&lds_vT[0][ve * 32 + vc] = vv;
  }
  __syncthreads();

  for (int kb = 0; kb < L_DIM; kb += 32) {
    int cur = (kb >> 5) & 1;
    bool has_next = (kb + 32) < L_DIM;
    v8h knext, vnext;
    if (has_next) {  // prefetch next tiles into registers
      knext = *(const v8h*)(ksrc + (long)(kb + 32) * DH);
      vnext = *(const v8h*)(vsrc + kb + 32);
    }

    // QK^T: load all four K fragments, fence, then the 4 WMMAs
    v16h bk0 = load_b_frag(&lds_k[cur][0], 64, 0, 0);
    v16h bk1 = load_b_frag(&lds_k[cur][0], 64, 0, 32);
    v16h bk2 = load_b_frag(&lds_k[cur][0], 64, 16, 0);
    v16h bk3 = load_b_frag(&lds_k[cur][0], 64, 16, 32);
    SCHED_FENCE();
    v8f att0 = (v8f)(0.0f), att1 = (v8f)(0.0f);
    att0 = wmma16(qa0, bk0, att0);
    att0 = wmma16(qa1, bk1, att0);
    att1 = wmma16(qa0, bk2, att1);
    att1 = wmma16(qa1, bk3, att1);
    SCHED_FENCE();

    // mask + online softmax (C layout: lane=col(key), vgpr r = row g*8+r)
#pragma unroll
    for (int r = 0; r < 8; r++) {
      int qrow = q0 + g * 8 + r;
      unsigned mw = Gbits[(long)qrow * LW + (kb >> 5)];
      float a0 = ((mw >> n16) & 1u) ? att0[r] * sc : NEG_BIG;
      float a1 = ((mw >> (16 + n16)) & 1u) ? att1[r] * sc : NEG_BIG;
      float mx = fmaxf(a0, a1);
#pragma unroll
      for (int off = 1; off < 16; off <<= 1)
        mx = fmaxf(mx, __shfl_xor(mx, off, 32));
      float mn = fmaxf(m_s[r], mx);
      float p0 = __expf(a0 - mn);
      float p1 = __expf(a1 - mn);
      float rs = p0 + p1;
#pragma unroll
      for (int off = 1; off < 16; off <<= 1) rs += __shfl_xor(rs, off, 32);
      float alpha = __expf(m_s[r] - mn);
      l_s[r] = l_s[r] * alpha + rs;
      m_s[r] = mn;
#pragma unroll
      for (int t = 0; t < 4; t++) o[t][r] *= alpha;
      lds_p[w][(g * 8 + r) * 32 + n16] = (half_t)p0;
      lds_p[w][(g * 8 + r) * 32 + 16 + n16] = (half_t)p1;
    }

    // P @ V: v fragments first (independent of lds_p), then P, fence, WMMAs
    v16h vb0 = load_b_frag(&lds_vT[cur][0], 32, 0, 0);
    v16h vb1 = load_b_frag(&lds_vT[cur][0], 32, 16, 0);
    v16h vb2 = load_b_frag(&lds_vT[cur][0], 32, 32, 0);
    v16h vb3 = load_b_frag(&lds_vT[cur][0], 32, 48, 0);
    v16h pa = load_a_frag(&lds_p[w][0], 32, 0, 0);
    SCHED_FENCE();
    o[0] = wmma16(pa, vb0, o[0]);
    o[1] = wmma16(pa, vb1, o[1]);
    o[2] = wmma16(pa, vb2, o[2]);
    o[3] = wmma16(pa, vb3, o[3]);
    SCHED_FENCE();

    if (has_next) {  // commit prefetched tiles to the other buffer
      *(v8h*)&lds_k[cur ^ 1][kr * 64 + kc] = knext;
      *(v8h*)&lds_vT[cur ^ 1][ve * 32 + vc] = vnext;
    }
    __syncthreads();
  }

  // epilogue: normalize, write into concat[:, 512 + h*64 + e]
  float inv[8];
#pragma unroll
  for (int r = 0; r < 8; r++) inv[r] = 1.0f / l_s[r];
#pragma unroll
  for (int t = 0; t < 4; t++) {
#pragma unroll
    for (int r = 0; r < 8; r++) {
      int row = b * L_DIM + q0 + g * 8 + r;
      int col = DH + h * E_DIM + t * 16 + n16;
      concatA[(long)row * DCAT + col] = (half_t)(o[t][r] * inv[r]);
    }
  }
}

// ---------------------------------------------------------------------------
// Launch
// ---------------------------------------------------------------------------
extern "C" void kernel_launch(void* const* d_in, const int* in_sizes, int n_in,
                              void* d_out, int out_size, void* d_ws,
                              size_t ws_size, hipStream_t stream) {
  const float* s    = (const float*)d_in[0];   // [4,2048,256]
  const int*   Gmat = (const int*)  d_in[1];   // [2048,2048]
  const float* W1   = (const float*)d_in[2];   // [256,512]
  const float* b1   = (const float*)d_in[3];   // [512]
  const float* Qw   = (const float*)d_in[4];   // [512,512]
  const float* Kw   = (const float*)d_in[5];   // [512,512]
  const float* W2   = (const float*)d_in[6];   // [1024,512]
  const float* b2   = (const float*)d_in[7];   // [512]
  float* out = (float*)d_out;                  // [4,2048,512]

  char* ws = (char*)d_ws;
  size_t off = 0;
  auto alloc = [&](size_t bytes) {
    char* p = ws + off;
    off += (bytes + 255) & ~(size_t)255;
    return p;
  };
  half_t*   sf16    = (half_t*)alloc((size_t)M_ROWS * DIN * 2);       // 4 MB
  half_t*   concatA = (half_t*)alloc((size_t)M_ROWS * DCAT * 2);      // 16 MB
  half_t*   shT     = (half_t*)alloc((size_t)B_DIM * DH * L_DIM * 2); // 8 MB
  half_t*   qf16    = (half_t*)alloc((size_t)M_ROWS * DH * 2);        // 8 MB
  half_t*   kf16    = (half_t*)alloc((size_t)M_ROWS * DH * 2);        // 8 MB
  unsigned* Gbits   = (unsigned*)alloc((size_t)L_DIM * LW * 4);       // 512 KB
  half_t*   W1T     = (half_t*)alloc((size_t)DIN * DH * 2);
  half_t*   QwT     = (half_t*)alloc((size_t)DH * DH * 2);
  half_t*   KwT     = (half_t*)alloc((size_t)DH * DH * 2);
  half_t*   W2T     = (half_t*)alloc((size_t)DCAT * DH * 2);
  (void)ws_size; (void)in_sizes; (void)n_in; (void)out_size;

  // 0) conversions + mask packing
  int ns = M_ROWS * DIN;
  cvt_f32_to_f16<<<(ns + 255) / 256, 256, 0, stream>>>(s, sf16, ns);
  cvt_transpose<<<(DIN * DH + 255) / 256, 256, 0, stream>>>(W1, W1T, DIN, DH);
  cvt_transpose<<<(DH * DH + 255) / 256, 256, 0, stream>>>(Qw, QwT, DH, DH);
  cvt_transpose<<<(DH * DH + 255) / 256, 256, 0, stream>>>(Kw, KwT, DH, DH);
  cvt_transpose<<<(DCAT * DH + 255) / 256, 256, 0, stream>>>(W2, W2T, DCAT, DH);
  int nwords = L_DIM * LW;
  pack_mask<<<(nwords + 255) / 256, 256, 0, stream>>>(Gmat, Gbits, nwords);

  dim3 blk(256);
  // 1) s1 = s@W1 + b1 -> concat[:, :512] (f16) and shT
  gemm_wmma_f16<<<dim3(M_ROWS / 128, DH / 128), blk, 0, stream>>>(
      sf16, DIN, W1T, DIN, b1, DIN, nullptr, 0, concatA, DCAT, shT);
  // 2) q = s1@Qw, k = s1@Kw
  gemm_wmma_f16<<<dim3(M_ROWS / 128, DH / 128), blk, 0, stream>>>(
      concatA, DCAT, QwT, DH, nullptr, DH, nullptr, 0, qf16, DH, nullptr);
  gemm_wmma_f16<<<dim3(M_ROWS / 128, DH / 128), blk, 0, stream>>>(
      concatA, DCAT, KwT, DH, nullptr, DH, nullptr, 0, kf16, DH, nullptr);
  // 3) fused masked-softmax attention -> concat[:, 512:]
  attn_wmma<<<dim3(L_DIM / 128, H_DIM, B_DIM), blk, 0, stream>>>(
      qf16, kf16, shT, Gbits, concatA);
  // 4) out = concat @ W2 + b2 (f32)
  gemm_wmma_f16<<<dim3(M_ROWS / 128, DH / 128), blk, 0, stream>>>(
      concatA, DCAT, W2T, DCAT, b2, DCAT, out, DH, nullptr, 0, nullptr);
}
